// GraspEvaluator_18253611008648
// MI455X (gfx1250) — compile-verified, matrix operand order checked by reading, not run on hardware
//
#include <hip/hip_runtime.h>
#include <hip/hip_bf16.h>

typedef _Float16 half_t;
typedef __attribute__((ext_vector_type(16))) _Float16 v16h;
typedef __attribute__((ext_vector_type(8)))  _Float16 v8h;
typedef __attribute__((ext_vector_type(8)))  float    v8f;
typedef __attribute__((ext_vector_type(4)))  unsigned u32x4;
typedef __attribute__((ext_vector_type(8)))  int      i32x8;
typedef __attribute__((ext_vector_type(4)))  int      i32x4;

#define BATCH 16
#define NPTS  1152
#define KNN   20
#define P2    (NPTS * KNN)   // 23040 positions per batch in conv2d stages
#define PT    8              // p-tiles per workgroup in k_gemm
#define BNEPS 1e-5f

#if defined(__has_builtin)
#if __has_builtin(__builtin_amdgcn_tensor_load_to_lds) && __has_builtin(__builtin_amdgcn_s_wait_tensorcnt)
#define USE_TDM 1
#endif
#endif
#ifndef USE_TDM
#define USE_TDM 0
#endif

// ---------------------------------------------------------------- utilities

__device__ __forceinline__ float activate(float y, int act) {
  if (act == 1) y = (y > 0.f) ? y : 0.2f * y;            // LeakyReLU(0.2)
  else if (act == 2) y = fmaxf(y, 0.f);                  // ReLU
  else if (act == 3) y = 1.f / (1.f + __expf(-y));       // sigmoid
  return y;
}

// K index inside a 16-half WMMA fragment (ISA 7.12.2, 16-bit A 16x32 layout)
__device__ __forceinline__ int kmap(int i, int h) {
  return i + ((i >= 8) ? 8 : 0) + 8 * h;
}

// ------------------------------------------------------- weight conversion

// f32 (O,C) -> f16 (Opad,Cpad), zero padded
__global__ void k_cvt_w(const float* __restrict__ W, half_t* __restrict__ D,
                        int O, int C, int Opad, int Cpad) {
  int t = blockIdx.x * blockDim.x + threadIdx.x;
  if (t >= Opad * Cpad) return;
  int o = t / Cpad, c = t % Cpad;
  float v = (o < O && c < C) ? W[(size_t)o * C + c] : 0.f;
  D[t] = (half_t)v;
}

// ------------------------------------------------------------ input concat

// build l0_xyz (B,N,3) f32 and l0_points/rot (B,4,N) f32
__global__ void k_concat(const float* __restrict__ pc, const float* __restrict__ gpc,
                         float* __restrict__ xyz, float* __restrict__ rot) {
  int t = blockIdx.x * blockDim.x + threadIdx.x;
  if (t >= BATCH * NPTS) return;
  int b = t / NPTS, n = t % NPTS;
  float x, y, z, lab;
  if (n < 1024) {
    const float* p = pc + ((size_t)b * 1024 + n) * 3;
    x = p[0]; y = p[1]; z = p[2]; lab = 1.f;
  } else {
    const float* p = gpc + ((size_t)b * 128 + (n - 1024)) * 3;
    x = p[0]; y = p[1]; z = p[2]; lab = 0.f;
  }
  float* o = xyz + (size_t)t * 3;
  o[0] = x; o[1] = y; o[2] = z;
  float* r = rot + (size_t)b * 4 * NPTS + n;
  r[0] = x; r[(size_t)NPTS] = y; r[(size_t)2 * NPTS] = z; r[(size_t)3 * NPTS] = lab;
}

// -------------------------------------------------------------------- kNN

// pairwise = -||xi - xj||^2 for 3-d points (B,N,3)
__global__ void k_pair3(const float* __restrict__ xyz, float* __restrict__ pair) {
  long long t = (long long)blockIdx.x * blockDim.x + threadIdx.x;
  long long tot = (long long)BATCH * NPTS * NPTS;
  if (t >= tot) return;
  int m = (int)(t % NPTS);
  long long r = t / NPTS;
  int n = (int)(r % NPTS);
  int b = (int)(r / NPTS);
  const float* a = xyz + ((size_t)b * NPTS + n) * 3;
  const float* c = xyz + ((size_t)b * NPTS + m) * 3;
  float dx = a[0] - c[0], dy = a[1] - c[1], dz = a[2] - c[2];
  pair[t] = -(dx * dx + dy * dy + dz * dz);
}

// per-point ||x||^2 for 64-channel f16 feature maps (channel-major, row stride NPTS)
__global__ void k_xx64(const half_t* __restrict__ X, size_t xbs, float* __restrict__ xx) {
  int t = blockIdx.x * blockDim.x + threadIdx.x;
  if (t >= BATCH * NPTS) return;
  int b = t / NPTS, n = t % NPTS;
  const half_t* Xb = X + (size_t)b * xbs;
  float s = 0.f;
  for (int c = 0; c < 64; ++c) {
    float v = (float)Xb[(size_t)c * NPTS + n];
    s += v * v;
  }
  xx[t] = s;
}

// pairwise = 2*X^T X - xx_n - xx_m via WMMA (K = 64), one 16x16 (n,m) tile per wave
__global__ void k_pair64(const half_t* __restrict__ X, size_t xbs,
                         const float* __restrict__ xx, float* __restrict__ pair) {
  int lane = threadIdx.x, h = lane >> 4, r = lane & 15;
  int n0 = blockIdx.x * 16, m0 = blockIdx.y * 16, b = blockIdx.z;
  const half_t* Xb = X + (size_t)b * xbs;
  v8f acc = {0.f, 0.f, 0.f, 0.f, 0.f, 0.f, 0.f, 0.f};
  for (int kb = 0; kb < 64; kb += 32) {
    v16h a, bb;
#pragma unroll
    for (int i = 0; i < 16; ++i) {
      int kk = kb + kmap(i, h);
      a[i]  = Xb[(size_t)kk * NPTS + n0 + r];
      bb[i] = Xb[(size_t)kk * NPTS + m0 + r];
    }
    acc = __builtin_amdgcn_wmma_f32_16x16x32_f16(false, a, false, bb, (short)0, acc,
                                                 false, false);
  }
  const float* xxb = xx + (size_t)b * NPTS;
  float* pb = pair + (size_t)b * NPTS * NPTS;
#pragma unroll
  for (int j = 0; j < 8; ++j) {
    int n = n0 + j + 8 * h, m = m0 + r;
    pb[(size_t)n * NPTS + m] = 2.f * acc[j] - xxb[n] - xxb[m];
  }
}

// top-20 (descending, ties -> lower index) per row of (B,N,N) score matrix
__global__ void k_topk(const float* __restrict__ pair, int* __restrict__ idx) {
  int t = blockIdx.x * blockDim.x + threadIdx.x;
  if (t >= BATCH * NPTS) return;
  const float* row = pair + (size_t)t * NPTS;
  float bv[KNN];
  int bi[KNN];
  for (int i = 0; i < KNN; ++i) { bv[i] = -3.4e38f; bi[i] = 0; }
  for (int m = 0; m < NPTS; ++m) {
    float v = row[m];
    if (v > bv[KNN - 1]) {
      int p = KNN - 1;
      while (p > 0 && bv[p - 1] < v) { bv[p] = bv[p - 1]; bi[p] = bi[p - 1]; --p; }
      bv[p] = v; bi[p] = m;
    }
  }
  int* o = idx + (size_t)t * KNN;
  for (int i = 0; i < KNN; ++i) o[i] = bi[i];
}

// ---------------------------------------------------------- graph features

// graph feature from 3-d points; 32 padded channels: [diff(3) | center(3) | rot(4)|0...]
__global__ void k_gf_pts(const float* __restrict__ xyz, const float* __restrict__ rot,
                         const int* __restrict__ idx, half_t* __restrict__ out, int useRot) {
  int t = blockIdx.x * blockDim.x + threadIdx.x;
  if (t >= BATCH * NPTS * KNN) return;
  int kk = t % KNN;
  int r = t / KNN;
  int n = r % NPTS, b = r / NPTS;
  int j = idx[t];
  const float* pn = xyz + ((size_t)b * NPTS + n) * 3;
  const float* pj = xyz + ((size_t)b * NPTS + j) * 3;
  half_t* o = out + (size_t)b * 32 * P2 + (size_t)n * KNN + kk;
#pragma unroll
  for (int c = 0; c < 3; ++c) {
    o[(size_t)c * P2]       = (half_t)(pj[c] - pn[c]);
    o[(size_t)(c + 3) * P2] = (half_t)pn[c];
  }
#pragma unroll
  for (int c = 6; c < 10; ++c) {
    float v = useRot ? rot[(size_t)b * 4 * NPTS + (size_t)(c - 6) * NPTS + n] : 0.f;
    o[(size_t)c * P2] = (half_t)v;
  }
#pragma unroll
  for (int c = 10; c < 32; ++c) o[(size_t)c * P2] = (half_t)0.f;
}

// graph feature from 64-channel f16 features -> 128 channels [diff(64) | center(64)]
__global__ void k_gf_feat(const half_t* __restrict__ X, size_t xbs,
                          const int* __restrict__ idx, half_t* __restrict__ out) {
  int t = blockIdx.x * blockDim.x + threadIdx.x;
  if (t >= BATCH * NPTS * KNN) return;
  int kk = t % KNN;
  int r = t / KNN;
  int n = r % NPTS, b = r / NPTS;
  int j = idx[t];
  const half_t* Xb = X + (size_t)b * xbs;
  half_t* o = out + (size_t)b * 128 * P2 + (size_t)n * KNN + kk;
  for (int c = 0; c < 64; ++c) {
    float xn = (float)Xb[(size_t)c * NPTS + n];
    float xj = (float)Xb[(size_t)c * NPTS + j];
    o[(size_t)c * P2]        = (half_t)(xj - xn);
    o[(size_t)(64 + c) * P2] = (half_t)xn;
  }
}

// -------------------- WMMA GEMM (1x1 conv) + BN + act, TDM-staged X tiles

// Y[b][o][p] = act(BN(sum_c W[o][c] * X[b][c][p]))
// Block = (32, OW): OW waves, one o-tile each, sharing one LDS X tile.
// Each block walks PT p-tiles; W fragments live in registers for all K chunks.
template <int C>
__global__ void k_gemm(const half_t* __restrict__ X, const half_t* __restrict__ W,
                       const float* __restrict__ bn, half_t* __restrict__ Y,
                       int P, int O, int act) {
  __shared__ half_t tile[192 * 16];           // X tile: C rows x 16 positions (<= 6 KB)
  constexpr int KC = C / 32;
  int lane = threadIdx.x, h = lane >> 4, r = lane & 15;
  int o0 = (blockIdx.y * blockDim.y + threadIdx.y) * 16;
  int b = blockIdx.z;
  const half_t* Xb = X + (size_t)b * C * P;
  half_t* Yb = Y + (size_t)b * O * P;

  // preload W fragments: per K chunk, each lane's 16 halves are two contiguous
  // 16B runs -> two b128 loads (kmap: i<8 -> kb+8h+i, i>=8 -> kb+16+8h+(i-8))
  v16h wf[KC];
  const v8h* wrow = (const v8h*)(W + (size_t)(o0 + r) * C);
#pragma unroll
  for (int kc = 0; kc < KC; ++kc) {
    v8h lo = wrow[kc * 4 + h];
    v8h hi = wrow[kc * 4 + 2 + h];
#pragma unroll
    for (int i = 0; i < 8; ++i) { wf[kc][i] = lo[i]; wf[kc][i + 8] = hi[i]; }
  }

  for (int t = 0; t < PT; ++t) {
    int p0 = (blockIdx.x * PT + t) * 16;
#if USE_TDM
    if (threadIdx.y == 0) {
      // Tensor Data Mover: 2D D# -- tile_dim0=16 elems (2B), tile_dim1=C rows,
      // tensor_dim0_stride=P; destination LDS row stride = 32B (contiguous tile).
      unsigned lds = (unsigned)(size_t)(void*)tile;
      unsigned long long ga = (unsigned long long)(size_t)(Xb + p0);
      u32x4 g0;
      g0[0] = 1u;                                          // count=1, user D#
      g0[1] = lds;                                         // lds_addr
      g0[2] = (unsigned)(ga & 0xffffffffu);                // global_addr[31:0]
      g0[3] = (unsigned)((ga >> 32) & 0x1ffffffu) | (2u << 30);  // addr[56:32] | type=2
      i32x8 g1;
      g1[0] = (int)(1u << 16);                             // data_size=1 (2 bytes)
      g1[1] = (int)(((unsigned)P & 0xffffu) << 16);        // tensor_dim0[15:0]
      g1[2] = (int)((((unsigned)P >> 16) & 0xffffu) | (((unsigned)C & 0xffffu) << 16));
      g1[3] = (int)((((unsigned)C >> 16) & 0xffffu) | (16u << 16));   // tile_dim0=16
      g1[4] = (int)((unsigned)C & 0xffffu);                // tile_dim1=C, tile_dim2=0
      g1[5] = (int)(unsigned)P;                            // tensor_dim0_stride[31:0]
      g1[6] = 0;                                           // stride hi, dim1_stride lo
      g1[7] = 0;
      i32x4 z4 = {0, 0, 0, 0};
      i32x8 z8 = {0, 0, 0, 0, 0, 0, 0, 0};
      __builtin_amdgcn_tensor_load_to_lds(g0, g1, z4, z4, z8, 0);
      __builtin_amdgcn_s_wait_tensorcnt(0);
    }
#else
    {
      const half_t* src = Xb + p0;
      int tot = C * 2;                                     // 8-half units
      for (int u = threadIdx.y * 32 + lane; u < tot; u += blockDim.y * 32) {
        int row = u >> 1, part = u & 1;
        ((v8h*)tile)[u] = *(const v8h*)(src + (size_t)row * P + 8 * part);
      }
    }
#endif
    __syncthreads();

    v8f acc = {0.f, 0.f, 0.f, 0.f, 0.f, 0.f, 0.f, 0.f};
#pragma unroll
    for (int kc = 0; kc < KC; ++kc) {
      v16h bf;
#pragma unroll
      for (int i = 0; i < 16; ++i) bf[i] = tile[(kc * 32 + kmap(i, h)) * 16 + r];
      acc = __builtin_amdgcn_wmma_f32_16x16x32_f16(false, wf[kc], false, bf, (short)0,
                                                   acc, false, false);
    }
#pragma unroll
    for (int j = 0; j < 8; ++j) {
      int o = o0 + j + 8 * h;
      float y = acc[j];
      if (bn) y = (y - bn[2 * O + o]) * (bn[o] * rsqrtf(bn[3 * O + o] + BNEPS)) + bn[O + o];
      y = activate(y, act);
      Yb[(size_t)o * P + p0 + r] = (half_t)y;
    }
    __syncthreads();
  }
}

// ---------------------------------------------- WMMA fully-connected (M = 16 batch)

// Y(16,Opad) = act(BN(A(16,C) @ W(Opad,C)^T + bias)); f16 out and/or f32 out
__global__ void k_fc(const half_t* __restrict__ A, const half_t* __restrict__ W,
                     const float* __restrict__ bias, const float* __restrict__ bn,
                     half_t* __restrict__ Y16, float* __restrict__ Yf,
                     int C, int Opad, int Otrue, int act, int fstore) {
  int lane = threadIdx.x, h = lane >> 4, r = lane & 15;
  int o0 = blockIdx.x * 16;
  v8f acc = {0.f, 0.f, 0.f, 0.f, 0.f, 0.f, 0.f, 0.f};
  for (int kb = 0; kb < C; kb += 32) {
    v16h a, bb;
#pragma unroll
    for (int i = 0; i < 16; ++i) {
      int kk = kb + kmap(i, h);
      a[i]  = A[(size_t)r * C + kk];            // A rows = batch
      bb[i] = W[(size_t)(o0 + r) * C + kk];     // B cols = W^T columns = W rows
    }
    acc = __builtin_amdgcn_wmma_f32_16x16x32_f16(false, a, false, bb, (short)0, acc,
                                                 false, false);
  }
#pragma unroll
  for (int j = 0; j < 8; ++j) {
    int m = j + 8 * h, o = o0 + r;
    float y = acc[j];
    if (o < Otrue) {
      if (bias) y += bias[o];
      if (bn) y = (y - bn[2 * Otrue + o]) * (bn[o] * rsqrtf(bn[3 * Otrue + o] + BNEPS))
                  + bn[Otrue + o];
    }
    y = activate(y, act);
    if (Y16) Y16[(size_t)m * Opad + o] = (half_t)y;
    if (Yf && o < fstore) Yf[(size_t)m * fstore + o] = y;
  }
}

// ------------------------------------------------------------- reductions

// max over k (20) neighbors; write into dst at channel offset (f16)
__global__ void k_maxk(const half_t* __restrict__ S, half_t* __restrict__ D,
                       int C, int dstCtot, int coff) {
  int t = blockIdx.x * blockDim.x + threadIdx.x;
  if (t >= BATCH * C * NPTS) return;
  int n = t % NPTS;
  int r = t / NPTS;
  int c = r % C, b = r / C;
  const half_t* s = S + (size_t)b * C * P2 + (size_t)c * P2 + (size_t)n * KNN;
  float m = -3.4e38f;
#pragma unroll
  for (int i = 0; i < KNN; ++i) m = fmaxf(m, (float)s[i]);
  D[((size_t)b * dstCtot + coff + c) * NPTS + n] = (half_t)m;
}

// max over N points -> (B, C) row-major f16
__global__ void k_maxn(const half_t* __restrict__ S, half_t* __restrict__ D, int C) {
  int t = blockIdx.x * blockDim.x + threadIdx.x;
  if (t >= BATCH * C) return;
  const half_t* s = S + (size_t)t * NPTS;
  float m = -3.4e38f;
  for (int n = 0; n < NPTS; ++n) m = fmaxf(m, (float)s[n]);
  D[t] = (half_t)m;
}

// --------------------------------------------------------------- 3x3 bmm

__global__ void k_bmm(const float* __restrict__ pts, const float* __restrict__ tmat,
                      float* __restrict__ out) {
  int t = blockIdx.x * blockDim.x + threadIdx.x;
  if (t >= BATCH * NPTS) return;
  int b = t / NPTS;
  const float* p = pts + (size_t)t * 3;
  const float* T = tmat + (size_t)b * 16;   // row-major 3x3 in cols 0..8
  float* o = out + (size_t)t * 3;
#pragma unroll
  for (int d = 0; d < 3; ++d)
    o[d] = p[0] * T[d] + p[1] * T[3 + d] + p[2] * T[6 + d];
}

// ================================================================ launch

extern "C" void kernel_launch(void* const* d_in, const int* in_sizes, int n_in,
                              void* d_out, int out_size, void* d_ws, size_t ws_size,
                              hipStream_t stream) {
  const float* pc   = (const float*)d_in[0];
  const float* gpc  = (const float*)d_in[1];
  const float* p_tc1w = (const float*)d_in[2];  const float* p_tc1b = (const float*)d_in[3];
  const float* p_tc2w = (const float*)d_in[4];  const float* p_tc2b = (const float*)d_in[5];
  const float* p_tc3w = (const float*)d_in[6];  const float* p_tc3b = (const float*)d_in[7];
  const float* p_tl1w = (const float*)d_in[8];  const float* p_tl1b = (const float*)d_in[9];
  const float* p_tl2w = (const float*)d_in[10]; const float* p_tl2b = (const float*)d_in[11];
  const float* p_ttfw = (const float*)d_in[12]; const float* p_ttfb = (const float*)d_in[13];
  const float* p_c1w = (const float*)d_in[14];  const float* p_c1b = (const float*)d_in[15];
  const float* p_c2w = (const float*)d_in[16];  const float* p_c2b = (const float*)d_in[17];
  const float* p_c3w = (const float*)d_in[18];  const float* p_c3b = (const float*)d_in[19];
  const float* p_c4w = (const float*)d_in[20];  const float* p_c4b = (const float*)d_in[21];
  const float* p_c5w = (const float*)d_in[22];  const float* p_c5b = (const float*)d_in[23];
  const float* p_c6w = (const float*)d_in[24];  const float* p_c6b = (const float*)d_in[25];
  const float* p_f1w = (const float*)d_in[26];  const float* p_f1bias = (const float*)d_in[27];
  const float* p_f1bn = (const float*)d_in[28];
  const float* p_f2w = (const float*)d_in[29];  const float* p_f2bias = (const float*)d_in[30];
  const float* p_f2bn = (const float*)d_in[31];
  const float* p_lw = (const float*)d_in[32];   const float* p_lb = (const float*)d_in[33];
  const float* p_cw = (const float*)d_in[34];   const float* p_cb = (const float*)d_in[35];

  char* ws = (char*)d_ws;
  size_t off = 0;
  auto alloc = [&](size_t bytes) -> char* {
    char* p = ws + off;
    off += (bytes + 255) & ~(size_t)255;
    return p;
  };

  // converted f16 weights (padded)
  half_t* w_tc1 = (half_t*)alloc((size_t)64 * 32 * 2);
  half_t* w_tc2 = (half_t*)alloc((size_t)128 * 64 * 2);
  half_t* w_tc3 = (half_t*)alloc((size_t)1024 * 128 * 2);
  half_t* w_tl1 = (half_t*)alloc((size_t)512 * 1024 * 2);
  half_t* w_tl2 = (half_t*)alloc((size_t)256 * 512 * 2);
  half_t* w_ttf = (half_t*)alloc((size_t)16 * 256 * 2);
  half_t* w_c1  = (half_t*)alloc((size_t)64 * 32 * 2);
  half_t* w_c2  = (half_t*)alloc((size_t)64 * 64 * 2);
  half_t* w_c3  = (half_t*)alloc((size_t)64 * 128 * 2);
  half_t* w_c4  = (half_t*)alloc((size_t)64 * 64 * 2);
  half_t* w_c5  = (half_t*)alloc((size_t)64 * 128 * 2);
  half_t* w_c6  = (half_t*)alloc((size_t)512 * 192 * 2);
  half_t* w_f1  = (half_t*)alloc((size_t)1024 * 512 * 2);
  half_t* w_f2  = (half_t*)alloc((size_t)1024 * 1024 * 2);
  half_t* w_lg  = (half_t*)alloc((size_t)16 * 1024 * 2);
  half_t* w_cf  = (half_t*)alloc((size_t)16 * 1024 * 2);

  float*  xyz   = (float*)alloc((size_t)BATCH * NPTS * 3 * 4);
  float*  rot   = (float*)alloc((size_t)BATCH * 4 * NPTS * 4);
  float*  xform = (float*)alloc((size_t)BATCH * NPTS * 3 * 4);
  int*    idx   = (int*)alloc((size_t)BATCH * NPTS * KNN * 4);
  float*  xx    = (float*)alloc((size_t)BATCH * NPTS * 4);
  half_t* tb128 = (half_t*)alloc((size_t)BATCH * 128 * NPTS * 2);
  half_t* xcat  = (half_t*)alloc((size_t)BATCH * 192 * NPTS * 2);
  half_t* tfeat = (half_t*)alloc((size_t)16 * 1024 * 2);
  half_t* fbuf1 = (half_t*)alloc((size_t)16 * 1024 * 2);
  half_t* fbuf2 = (half_t*)alloc((size_t)16 * 1024 * 2);
  half_t* fbuf3 = (half_t*)alloc((size_t)16 * 1024 * 2);
  float*  tmat  = (float*)alloc((size_t)16 * 16 * 4);
  char*   regA  = alloc((size_t)BATCH * 128 * P2 * 2);   // 94.4 MB (aliases f32 pair matrix)
  char*   regB  = alloc((size_t)BATCH * 64 * P2 * 2);    // 47.2 MB

  half_t* A16 = (half_t*)regA;
  half_t* B16 = (half_t*)regB;
  float*  pairA = (float*)regA;   // B*N*N f32 = 84.9 MB <= regA

  auto blk = [](long long n) { return dim3((unsigned)((n + 255) / 256)); };
  const dim3 T256(256);
  const size_t xcat_bs = (size_t)192 * NPTS;

  // templated GEMM dispatch: OW waves share an LDS X tile; PT p-tiles per block
  auto gemm = [&](const half_t* Xp, const half_t* Wp, const float* bnp, half_t* Yp,
                  int Cc, int Pp, int Oo, int actv) {
    int OW = (Oo >= 128) ? 8 : Oo / 16;
    dim3 g(Pp / (16 * PT), Oo / (16 * OW), BATCH), bdim(32, OW);
    switch (Cc) {
      case 32:  k_gemm<32><<<g, bdim, 0, stream>>>(Xp, Wp, bnp, Yp, Pp, Oo, actv); break;
      case 64:  k_gemm<64><<<g, bdim, 0, stream>>>(Xp, Wp, bnp, Yp, Pp, Oo, actv); break;
      case 128: k_gemm<128><<<g, bdim, 0, stream>>>(Xp, Wp, bnp, Yp, Pp, Oo, actv); break;
      default:  k_gemm<192><<<g, bdim, 0, stream>>>(Xp, Wp, bnp, Yp, Pp, Oo, actv); break;
    }
  };

  // --- weight conversion
  k_cvt_w<<<blk(64 * 32), T256, 0, stream>>>(p_tc1w, w_tc1, 64, 6, 64, 32);
  k_cvt_w<<<blk(128 * 64), T256, 0, stream>>>(p_tc2w, w_tc2, 128, 64, 128, 64);
  k_cvt_w<<<blk(1024 * 128), T256, 0, stream>>>(p_tc3w, w_tc3, 1024, 128, 1024, 128);
  k_cvt_w<<<blk(512 * 1024), T256, 0, stream>>>(p_tl1w, w_tl1, 512, 1024, 512, 1024);
  k_cvt_w<<<blk(256 * 512), T256, 0, stream>>>(p_tl2w, w_tl2, 256, 512, 256, 512);
  k_cvt_w<<<blk(16 * 256), T256, 0, stream>>>(p_ttfw, w_ttf, 9, 256, 16, 256);
  k_cvt_w<<<blk(64 * 32), T256, 0, stream>>>(p_c1w, w_c1, 64, 10, 64, 32);
  k_cvt_w<<<blk(64 * 64), T256, 0, stream>>>(p_c2w, w_c2, 64, 64, 64, 64);
  k_cvt_w<<<blk(64 * 128), T256, 0, stream>>>(p_c3w, w_c3, 64, 128, 64, 128);
  k_cvt_w<<<blk(64 * 64), T256, 0, stream>>>(p_c4w, w_c4, 64, 64, 64, 64);
  k_cvt_w<<<blk(64 * 128), T256, 0, stream>>>(p_c5w, w_c5, 64, 128, 64, 128);
  k_cvt_w<<<blk(512 * 192), T256, 0, stream>>>(p_c6w, w_c6, 512, 192, 512, 192);
  k_cvt_w<<<blk(1024 * 512), T256, 0, stream>>>(p_f1w, w_f1, 1024, 512, 1024, 512);
  k_cvt_w<<<blk(1024 * 1024), T256, 0, stream>>>(p_f2w, w_f2, 1024, 1024, 1024, 1024);
  k_cvt_w<<<blk(16 * 1024), T256, 0, stream>>>(p_lw, w_lg, 1, 1024, 16, 1024);
  k_cvt_w<<<blk(16 * 1024), T256, 0, stream>>>(p_cw, w_cf, 1, 1024, 16, 1024);

  // --- inputs
  k_concat<<<blk(BATCH * NPTS), T256, 0, stream>>>(pc, gpc, xyz, rot);

  const dim3 W32(32);
  const long long BNK = (long long)BATCH * NPTS * KNN;

  // ===== transform net =====
  k_pair3<<<blk((long long)BATCH * NPTS * NPTS), T256, 0, stream>>>(xyz, pairA);
  k_topk<<<blk(BATCH * NPTS), T256, 0, stream>>>(pairA, idx);
  k_gf_pts<<<blk(BNK), T256, 0, stream>>>(xyz, rot, idx, A16, 0);
  gemm(A16, w_tc1, p_tc1b, B16, 32, P2, 64, 1);
  gemm(B16, w_tc2, p_tc2b, A16, 64, P2, 128, 1);
  k_maxk<<<blk((long long)BATCH * 128 * NPTS), T256, 0, stream>>>(A16, tb128, 128, 128, 0);
  gemm(tb128, w_tc3, p_tc3b, B16, 128, NPTS, 1024, 1);
  k_maxn<<<blk(BATCH * 1024), T256, 0, stream>>>(B16, tfeat, 1024);
  k_fc<<<dim3(512 / 16), W32, 0, stream>>>(tfeat, w_tl1, nullptr, p_tl1b, fbuf1, nullptr, 1024, 512, 512, 1, 0);
  k_fc<<<dim3(256 / 16), W32, 0, stream>>>(fbuf1, w_tl2, nullptr, p_tl2b, fbuf2, nullptr, 512, 256, 256, 1, 0);
  k_fc<<<dim3(1), W32, 0, stream>>>(fbuf2, w_ttf, p_ttfb, nullptr, nullptr, tmat, 256, 16, 9, 0, 16);
  k_bmm<<<blk(BATCH * NPTS), T256, 0, stream>>>(xyz, tmat, xform);

  // ===== stage 1: graph feature on transformed points (+rot) -> c1,c2 -> x1 =====
  k_pair3<<<blk((long long)BATCH * NPTS * NPTS), T256, 0, stream>>>(xform, pairA);
  k_topk<<<blk(BATCH * NPTS), T256, 0, stream>>>(pairA, idx);
  k_gf_pts<<<blk(BNK), T256, 0, stream>>>(xform, rot, idx, A16, 1);
  gemm(A16, w_c1, p_c1b, B16, 32, P2, 64, 1);
  gemm(B16, w_c2, p_c2b, A16, 64, P2, 64, 1);
  k_maxk<<<blk((long long)BATCH * 64 * NPTS), T256, 0, stream>>>(A16, xcat, 64, 192, 0);

  // ===== stage 2: kNN in 64-d (WMMA) -> c3,c4 -> x2 =====
  k_xx64<<<blk(BATCH * NPTS), T256, 0, stream>>>(xcat, xcat_bs, xx);
  k_pair64<<<dim3(NPTS / 16, NPTS / 16, BATCH), W32, 0, stream>>>(xcat, xcat_bs, xx, pairA);
  k_topk<<<blk(BATCH * NPTS), T256, 0, stream>>>(pairA, idx);
  k_gf_feat<<<blk(BNK), T256, 0, stream>>>(xcat, xcat_bs, idx, A16);
  gemm(A16, w_c3, p_c3b, B16, 128, P2, 64, 1);
  gemm(B16, w_c4, p_c4b, A16, 64, P2, 64, 1);
  k_maxk<<<blk((long long)BATCH * 64 * NPTS), T256, 0, stream>>>(A16, xcat, 64, 192, 64);

  // ===== stage 3: kNN in 64-d on x2 -> c5 -> x3 =====
  k_xx64<<<blk(BATCH * NPTS), T256, 0, stream>>>(xcat + (size_t)64 * NPTS, xcat_bs, xx);
  k_pair64<<<dim3(NPTS / 16, NPTS / 16, BATCH), W32, 0, stream>>>(xcat + (size_t)64 * NPTS, xcat_bs, xx, pairA);
  k_topk<<<blk(BATCH * NPTS), T256, 0, stream>>>(pairA, idx);
  k_gf_feat<<<blk(BNK), T256, 0, stream>>>(xcat + (size_t)64 * NPTS, xcat_bs, idx, A16);
  gemm(A16, w_c5, p_c5b, B16, 128, P2, 64, 1);
  k_maxk<<<blk((long long)BATCH * 64 * NPTS), T256, 0, stream>>>(B16, xcat, 64, 192, 128);

  // ===== c6 + global max + FC head =====
  gemm(xcat, w_c6, p_c6b, B16, 192, NPTS, 512, 1);
  k_maxn<<<blk(BATCH * 512), T256, 0, stream>>>(B16, fbuf3, 512);
  k_fc<<<dim3(1024 / 16), W32, 0, stream>>>(fbuf3, w_f1, p_f1bias, p_f1bn, fbuf1, nullptr, 512, 1024, 1024, 2, 0);
  k_fc<<<dim3(1024 / 16), W32, 0, stream>>>(fbuf1, w_f2, p_f2bias, p_f2bn, fbuf2, nullptr, 1024, 1024, 1024, 2, 0);
  // logits -> d_out[0..15], conf (sigmoid) -> d_out[16..31]
  k_fc<<<dim3(1), W32, 0, stream>>>(fbuf2, w_lg, p_lb, nullptr, nullptr, (float*)d_out, 1024, 16, 1, 0, 1);
  k_fc<<<dim3(1), W32, 0, stream>>>(fbuf2, w_cf, p_cb, nullptr, nullptr, (float*)d_out + 16, 1024, 16, 1, 3, 1);
  (void)in_sizes; (void)n_in; (void)out_size; (void)ws_size;
}
